// MyModel_16277926052613
// MI455X (gfx1250) — matrix operand order repaired and checked by reading, not compiled, not run on hardware
//
#include <hip/hip_runtime.h>
#include <hip/hip_bf16.h>
#include <math.h>

typedef __attribute__((ext_vector_type(16))) _Float16 v16h;
typedef __attribute__((ext_vector_type(8)))  float    v8f;

#define S 128
#define G3 384   // 3*S

// ---------------------------------------------------------------------------
// Kernel 1: fold the projection GEMM into the input GEMM.
//   Wc[j][k] = sum_{p<127} w_ih[j][p] * proj_w[p][k]   (stored fp16 for WMMA)
//   bc[j]    = b_ih[j] + sum_{p<127} w_ih[j][p] * proj_b[p]
// 49152 outputs, 127-MAC dots each: trivial cost, done in fp32.
// ---------------------------------------------------------------------------
__global__ void __launch_bounds__(256)
fold_kernel(const float* __restrict__ w_ih, const float* __restrict__ proj_w,
            const float* __restrict__ proj_b, const float* __restrict__ b_ih,
            _Float16* __restrict__ Wc16, float* __restrict__ bc) {
    int gid = blockIdx.x * blockDim.x + threadIdx.x;   // 0 .. 49151
    int j = gid >> 7;     // 0 .. 383 (output gate row)
    int k = gid & 127;    // 0 .. 127 (input feature)
    float acc = 0.f;
    #pragma unroll 1
    for (int p = 0; p < S - 1; ++p)
        acc = fmaf(w_ih[j * S + p], proj_w[p * S + k], acc);
    Wc16[j * S + k] = (_Float16)acc;
    if (k == 0) {
        float b = b_ih[j];
        for (int p = 0; p < S - 1; ++p)
            b = fmaf(w_ih[j * S + p], proj_b[p], b);
        bc[j] = b;
    }
}

// ---------------------------------------------------------------------------
// Kernel 2: gi = evs @ Wc^T + bc  via v_wmma_f32_16x16x32_f16.
// One wave per 16x16 output tile; K=128 in 4 chunks of 32.
// A (evs tile, fp32->fp16 on the fly): contiguous K = 16*hi + e per lane half, same order as B.
// B (Wc rows = K-contiguous): lanes 0-15 K=0-15, lanes 16-31 K=16-31.
// ---------------------------------------------------------------------------
__global__ void __launch_bounds__(256)
gi_gemm_kernel(const float* __restrict__ evs, const _Float16* __restrict__ Wc16,
               const float* __restrict__ bc, float* __restrict__ gi) {
    const int lane = threadIdx.x & 31;
    const int wave = threadIdx.x >> 5;
    const int tile = blockIdx.x * 8 + wave;   // grid covers mtiles*24 exactly
    const int mt = tile / 24;                 // M tile (rows of evs)
    const int nt = tile - mt * 24;            // N tile (gate outputs)
    const int r  = lane & 15;                 // A row / B col / C col
    const int hi = lane >> 4;

    const float*    arow = evs  + (size_t)(mt * 16 + r) * S + hi * 16;
    const _Float16* brow = Wc16 + (size_t)(nt * 16 + r) * S + hi * 16;

    v8f c = {};
    #pragma unroll
    for (int kc = 0; kc < 4; ++kc) {
        const int k0 = kc * 32;
        v16h a, b;
        #pragma unroll
        for (int e = 0; e < 8; ++e) {
            a[e]     = (_Float16)arow[k0 + e];        // K = k0 + 16*hi + e
            a[8 + e] = (_Float16)arow[k0 + 8 + e];    // K = k0 + 16*hi + 8 + e
        }
        b = *(const v16h*)(brow + k0);                // K = k0 + 16*hi .. +15 (32B aligned)
        c = __builtin_amdgcn_wmma_f32_16x16x32_f16(
                /*neg_a=*/false, a, /*neg_b=*/false, b,
                /*c_mod=*/(short)0, c, /*reuse_a=*/false, /*reuse_b=*/false);
    }

    const float bias = bc[nt * 16 + r];
    float* orow = gi + (size_t)(mt * 16 + hi * 8) * G3 + nt * 16 + r;
    #pragma unroll
    for (int v = 0; v < 8; ++v)                       // M = v + 8*hi, N = lane&15
        orow[(size_t)v * G3] = c[v] + bias;
}

// ---------------------------------------------------------------------------
// Kernel 3: the sequential GRU scan + final head. One workgroup, 384 threads
// (12 wave32s). w_hh resident in LDS (384 x 132 fp32, ~198KB of the 320KB
// WGP LDS). h double-buffered in LDS; own h element also kept in a register.
// gi streamed from the 192MB L2 with a one-step-ahead prefetch.
// 2 barriers per step.
// ---------------------------------------------------------------------------
#define WSTRIDE 132   // 4*d mod 64 != 0 for lane deltas d=1..15,17..31

__global__ void __launch_bounds__(G3, 1)
scan_kernel(const float* __restrict__ gi, const float* __restrict__ h0,
            const float* __restrict__ w_hh, const float* __restrict__ b_hh,
            const float* __restrict__ w_ih, const float* __restrict__ final_w,
            const float* __restrict__ final_b, float* __restrict__ out, int N) {
    extern __shared__ float sh[];
    float* sh_w  = sh;                       // 384 * 132
    float* sh_h  = sh_w + G3 * WSTRIDE;      // 2 * 128 (double buffer)
    float* sh_gh = sh_h + 2 * S;             // 384
    float* sh_gi = sh_gh + G3;               // 384
    const int tid = threadIdx.x;

    // Stage w_hh into padded LDS (coalesced global reads).
    for (int i = tid; i < G3 * S; i += G3) {
        int j = i >> 7, k = i & 127;
        sh_w[j * WSTRIDE + k] = w_hh[i];
    }
    float hj = 0.f;
    if (tid < S) { hj = h0[tid]; sh_h[tid] = hj; }
    const float bhh   = b_hh[tid];
    const float wmask = w_ih[tid * S + (S - 1)];  // mask column contribution (row 0 only)
    __syncthreads();

    const float* wrow = sh_w + tid * WSTRIDE;
    for (int t = 0; t < N; ++t) {
        const float* grow = gi + (size_t)t * G3;
        float gval = grow[tid];                    // L2-resident stream, issued early
        if (t == 0) gval += wmask;
        if (t + 1 < N)
            __builtin_prefetch(grow + G3 + tid, 0, 1);  // global_prefetch_b8, next step

        const float* hbuf = sh_h + (t & 1) * S;
        float acc = bhh;
        #pragma unroll 8
        for (int k = 0; k < S; k += 4) {           // b128 LDS reads; h is broadcast
            float4 wv = *(const float4*)(wrow + k);
            float4 hv = *(const float4*)(hbuf + k);
            acc = fmaf(wv.x, hv.x, acc);
            acc = fmaf(wv.y, hv.y, acc);
            acc = fmaf(wv.z, hv.z, acc);
            acc = fmaf(wv.w, hv.w, acc);
        }
        sh_gh[tid] = acc;
        sh_gi[tid] = gval;
        __syncthreads();

        if (tid < S) {
            float r  = 1.f / (1.f + expf(-(sh_gi[tid]         + sh_gh[tid])));
            float z  = 1.f / (1.f + expf(-(sh_gi[S + tid]     + sh_gh[S + tid])));
            float nn = tanhf(sh_gi[2 * S + tid] + r * sh_gh[2 * S + tid]);
            hj = (1.f - z) * nn + z * hj;
            sh_h[((t + 1) & 1) * S + tid] = hj;    // write the *other* buffer: no WAR race
        }
        __syncthreads();
    }

    // Final head: out = h @ final_w^T + final_b  (3 outputs)
    if (tid < 3) {
        const float* hf = sh_h + (N & 1) * S;
        float o = final_b[tid];
        for (int k = 0; k < S; ++k)
            o = fmaf(final_w[tid * S + k], hf[k], o);
        out[tid] = o;
    }
}

// ---------------------------------------------------------------------------
extern "C" void kernel_launch(void* const* d_in, const int* in_sizes, int n_in,
                              void* d_out, int out_size, void* d_ws, size_t ws_size,
                              hipStream_t stream) {
    const float* evs     = (const float*)d_in[0];
    const float* h0      = (const float*)d_in[1];
    const float* w_ih    = (const float*)d_in[2];
    const float* w_hh    = (const float*)d_in[3];
    const float* b_ih    = (const float*)d_in[4];
    const float* b_hh    = (const float*)d_in[5];
    const float* proj_w  = (const float*)d_in[6];
    const float* proj_b  = (const float*)d_in[7];
    const float* final_w = (const float*)d_in[8];
    const float* final_b = (const float*)d_in[9];
    float* out = (float*)d_out;
    const int N = in_sizes[0] / S;   // 16384

    // Workspace layout (all fully rewritten every call):
    _Float16* Wc16 = (_Float16*)d_ws;                      //  96 KB
    float*    bc   = (float*)((char*)d_ws + 98304);        // 1.5 KB
    float*    gi   = (float*)((char*)d_ws + 99840);        // N*384*4 = 24 MB

    fold_kernel<<<(G3 * S) / 256, 256, 0, stream>>>(w_ih, proj_w, proj_b, b_ih, Wc16, bc);

    const int mtiles = N / 16;
    gi_gemm_kernel<<<(mtiles * 24) / 8, 256, 0, stream>>>(evs, Wc16, bc, gi);

    const size_t lds_bytes = (size_t)(G3 * WSTRIDE + 2 * S + G3 + G3) * sizeof(float);
    scan_kernel<<<1, G3, lds_bytes, stream>>>(gi, h0, w_hh, b_hh, w_ih,
                                              final_w, final_b, out, N);
}
